// MaskedMultiHeadAttention_90709709291709
// MI455X (gfx1250) — matrix operand order repaired and checked by reading, not compiled, not run on hardware
//
#include <hip/hip_runtime.h>
#include <hip/hip_bf16.h>

typedef __attribute__((ext_vector_type(16))) _Float16 v16h;
typedef __attribute__((ext_vector_type(8)))  _Float16 v8h;
typedef __attribute__((ext_vector_type(4)))  _Float16 v4h;
typedef __attribute__((ext_vector_type(8)))  float    v8f;
typedef __attribute__((ext_vector_type(4)))  float    v4f;
typedef __attribute__((ext_vector_type(4)))  int      v4i;

#define B_  8
#define S_  1024
#define D_  1024
#define H_  16
#define DK_ 64
#define M_  (B_ * S_)   // 8192 rows

#define AS1 __attribute__((address_space(1)))
#define AS3 __attribute__((address_space(3)))

// CDNA5 async load-to-LDS path (ASYNCcnt-tracked), guarded per-toolchain.
#if defined(__has_builtin)
#if __has_builtin(__builtin_amdgcn_global_load_async_to_lds_b128) && \
    __has_builtin(__builtin_amdgcn_s_wait_asynccnt)
#define HAVE_ASYNC_LDS 1
#endif
#endif

__device__ __forceinline__ void async_b128(const _Float16* g, _Float16* l) {
#if defined(HAVE_ASYNC_LDS)
  __builtin_amdgcn_global_load_async_to_lds_b128((AS1 v4i*)g, (AS3 v4i*)l, 0, 0);
#else
  *(v8h*)l = *(const v8h*)g;
#endif
}
__device__ __forceinline__ void async_wait() {
#if defined(HAVE_ASYNC_LDS)
  __builtin_amdgcn_s_wait_asynccnt(0);
#endif
}

// ---------------------------------------------------------------------------
// WMMA helper: D = A(16x32 f16) * B(32x16 f16) + C(16x16 f32)
// ---------------------------------------------------------------------------
__device__ __forceinline__ v8f wmma_f16(v16h a, v16h b, v8f c) {
  return __builtin_amdgcn_wmma_f32_16x16x32_f16(false, a, false, b, (short)0, c,
                                                false, false);
}

// ---------------------------------------------------------------------------
// Kernel 0: f32 -> f16 elementwise convert (vectorized x4)
// ---------------------------------------------------------------------------
__global__ __launch_bounds__(256) void k_cvt_f16(const float* __restrict__ in,
                                                 _Float16* __restrict__ out,
                                                 int n4) {
  int i = blockIdx.x * blockDim.x + threadIdx.x;
  if (i >= n4) return;
  int base = i * 4;
  float4 v = *(const float4*)(in + base);
  v4h h;
  h[0] = (_Float16)v.x; h[1] = (_Float16)v.y;
  h[2] = (_Float16)v.z; h[3] = (_Float16)v.w;
  *(v4h*)(out + base) = h;
}

// ---------------------------------------------------------------------------
// Kernel 1: weight transpose + convert:  wt[n*D + k] = (f16) w[k*D + n]
// ---------------------------------------------------------------------------
__global__ void k_wtrans(const float* __restrict__ w, _Float16* __restrict__ wt) {
  __shared__ float tile[32][33];
  int bx = blockIdx.x * 32;   // n base
  int by = blockIdx.y * 32;   // k base
  int tx = threadIdx.x;       // 0..31
  int ty = threadIdx.y;       // 0..7
#pragma unroll
  for (int r = 0; r < 32; r += 8)
    tile[ty + r][tx] = w[(size_t)(by + ty + r) * D_ + (bx + tx)];
  __syncthreads();
#pragma unroll
  for (int r = 0; r < 32; r += 8)
    wt[(size_t)(bx + ty + r) * D_ + (by + tx)] = (_Float16)tile[tx][ty + r];
}

// ---------------------------------------------------------------------------
// Kernel 2: GEMM  C[M,N] = A[M,K](f16) * Wt[N,K](f16)^T + bias (f32)
// Block: 256 threads = 8 waves; block tile 128(M) x 64(N).
// B tile [64 x 32] double-buffered in LDS via async-to-LDS: the next tile's
// transfer is in flight while WMMAs consume the current one (1 barrier/step).
// Rows padded to 72 halves -> conflict-free ds fragment reads.
// ---------------------------------------------------------------------------
#define BT_STRIDE 72
template <bool F16OUT>
__global__ __launch_bounds__(256) void k_gemm(const _Float16* __restrict__ A,
                                              const _Float16* __restrict__ Wt,
                                              const float* __restrict__ bias,
                                              void* __restrict__ Cout) {
  const int K = D_, N = D_;
  __shared__ _Float16 Bt[2][64 * BT_STRIDE];
  const int lane = threadIdx.x & 31;
  const int wave = threadIdx.x >> 5;
  const int l15  = lane & 15;
  const int hi   = lane >> 4;               // 0 or 1 (half-wave)
  const int mbase = blockIdx.y * 128 + wave * 16;
  const int nbase = blockIdx.x * 64;

  v8f acc[4];
#pragma unroll
  for (int t = 0; t < 4; ++t) {
    float bv = bias[nbase + t * 16 + l15];
#pragma unroll
    for (int r = 0; r < 8; ++r) acc[t][r] = bv;
  }

  // Per-thread staging slot: 16 B of the 64x32-half B tile.
  const int brow = threadIdx.x >> 2;        // 0..63 (column n within slab)
  const int bseg = threadIdx.x & 3;         // which 8-half segment
  const _Float16* bsrc = Wt + (size_t)(nbase + brow) * K + bseg * 8;
  const int bslot = brow * BT_STRIDE + bseg * 8;

  const _Float16* arow = A + (size_t)(mbase + l15) * K;

  async_b128(bsrc, &Bt[0][bslot]);          // prologue: stage tile 0
  for (int kc = 0; kc < K; kc += 32) {
    const int ib = (kc >> 5) & 1;
    async_wait();                           // previous issue complete
    __syncthreads();                        // tile published block-wide
    if (kc + 32 < K)                        // stage next tile while computing
      async_b128(bsrc + kc + 32, &Bt[ib ^ 1][bslot]);

    // A fragment: lanes 0-15 row m=l15 hold K=0..15; lanes 16-31 hold K=16..31
    union { v16h v; v8h h[2]; } af;
    const _Float16* ap = arow + kc + hi * 16;
    af.h[0] = *(const v8h*)(ap);
    af.h[1] = *(const v8h*)(ap + 8);
    if (kc + 64 < K) __builtin_prefetch(arow + kc + 64, 0, 0);

    v16h bf[4];
#pragma unroll
    for (int t = 0; t < 4; ++t)
      bf[t] = *(const v16h*)(&Bt[ib][(t * 16 + l15) * BT_STRIDE + hi * 16]);
#pragma unroll
    for (int t = 0; t < 4; ++t)
      acc[t] = wmma_f16(af.v, bf[t], acc[t]);
  }

#pragma unroll
  for (int t = 0; t < 4; ++t)
#pragma unroll
    for (int r = 0; r < 8; ++r) {
      int row = mbase + r + hi * 8;
      int col = nbase + t * 16 + l15;
      if (F16OUT)
        ((_Float16*)Cout)[(size_t)row * N + col] = (_Float16)acc[t][r];
      else
        ((float*)Cout)[(size_t)row * N + col] = acc[t][r];
    }
}

// ---------------------------------------------------------------------------
// Kernel 3: causal flash attention per (b,h), 64 query rows per block.
// Block: 128 threads = 4 waves; wave owns 16 query rows.
//   - K tile [32 keys x 64 dk] + transposed V tile double-buffered; K via
//     async-to-LDS, next key block staged while current one is computed.
//   - softmax: scores scatter to LDS (f32), each lane owns one row half ->
//     local reduce + one shfl_xor(16) instead of per-row bpermute trees.
// ---------------------------------------------------------------------------
#define KT_STRIDE 72
#define VT_STRIDE 40
#define PT_STRIDE 40
#define SF_STRIDE 36
__global__ __launch_bounds__(128) void k_attn(const _Float16* __restrict__ Qh,
                                              const _Float16* __restrict__ Kh,
                                              const _Float16* __restrict__ Vh,
                                              _Float16* __restrict__ Ctx) {
  const int b  = blockIdx.x / H_;
  const int h  = blockIdx.x % H_;
  const int qb = blockIdx.y;              // 64-row query block
  const int lane = threadIdx.x & 31;
  const int wave = threadIdx.x >> 5;
  const int l15  = lane & 15;
  const int hi   = lane >> 4;

  const size_t bS   = (size_t)b * S_;
  const int    hoff = h * DK_;

  __shared__ _Float16 Kt[2][32 * KT_STRIDE];    // [key][dk] K tiles
  __shared__ _Float16 Vt[2][64 * VT_STRIDE];    // [dk][key] transposed V tiles
  __shared__ _Float16 Pt[4][16 * PT_STRIDE];    // per-wave P tile (f16)
  __shared__ float    Sf[4][16 * SF_STRIDE];    // per-wave score tile (f32)

  // per-lane softmax row ownership: lane owns row (lane&15), half (lane>>4)
  float m_own = -3.0e38f, l_own = 0.0f;

  v8f acc[4] = {};

  // Q fragments for both 32-wide DK chunks (loaded once)
  union { v16h v; v8h h2[2]; } qf[2];
  const int qrow_a = qb * 64 + wave * 16 + l15;
#pragma unroll
  for (int kk = 0; kk < 2; ++kk) {
    const _Float16* qp = Qh + (bS + qrow_a) * D_ + hoff + kk * 32 + hi * 16;
    qf[kk].h2[0] = *(const v8h*)(qp);
    qf[kk].h2[1] = *(const v8h*)(qp + 8);
  }

  // staging assignments (128 threads)
  const int key_s = threadIdx.x >> 2;           // 0..31
  const int seg_s = threadIdx.x & 3;            // 0..3 (16 halves each)
  const _Float16* ksrc = Kh + (bS + key_s) * D_ + hoff + seg_s * 16;
  const int kslot = key_s * KT_STRIDE + seg_s * 16;
  const _Float16* vsrc = Vh + (bS + key_s) * D_ + hoff + seg_s * 16;

  const int jmax = qb * 2 + 1;            // causal: keys <= qb*64+63

  // stage key block j into buffer (j&1): K async, V transposed scatter
  auto stage = [&](int j) {
    const size_t joff = (size_t)j * 32 * D_;
    const int ibuf = j & 1;
    async_b128(ksrc + joff, &Kt[ibuf][kslot]);
    async_b128(ksrc + joff + 8, &Kt[ibuf][kslot + 8]);
    v16h vv = *(const v16h*)(vsrc + joff);
#pragma unroll
    for (int e = 0; e < 16; ++e)
      Vt[ibuf][(seg_s * 16 + e) * VT_STRIDE + key_s] = vv[e];
  };

  stage(0);                                // prologue
  for (int j = 0; j <= jmax; ++j) {
    const int kb = j * 32;
    const int ib = j & 1;

    async_wait();                          // previous stage's K complete
    __syncthreads();                       // tiles published block-wide
    if (j < jmax) stage(j + 1);            // overlap next stage with compute

    // ---- scores: S = Q @ K^T  (two 16-key N-tiles, contraction over DK=64)
    v8f sc[2];
#pragma unroll
    for (int t = 0; t < 2; ++t) {
      v8f z = {};
#pragma unroll
      for (int kk = 0; kk < 2; ++kk) {
        v16h kf = *(const v16h*)(&Kt[ib][(t * 16 + l15) * KT_STRIDE + kk * 32 + hi * 16]);
        z = wmma_f16(qf[kk].v, kf, z);
      }
      sc[t] = z;
    }

    // ---- scatter masked, scaled scores (C layout) to LDS f32 tile
    const float sscale = 0.125f;           // 1/sqrt(DK)
#pragma unroll
    for (int r = 0; r < 8; ++r) {
      int row = qb * 64 + wave * 16 + r + hi * 8;
      int c0 = kb + l15, c1 = kb + 16 + l15;
      float a0 = (c0 <= row) ? sc[0][r] * sscale : -1.0e9f;
      float a1 = (c1 <= row) ? sc[1][r] * sscale : -1.0e9f;
      Sf[wave][(r + hi * 8) * SF_STRIDE + l15]      = a0;
      Sf[wave][(r + hi * 8) * SF_STRIDE + 16 + l15] = a1;
    }
    asm volatile("s_wait_dscnt 0x0" ::: "memory");

    // ---- row-wise softmax: lane owns row l15, columns [hi*16, hi*16+16)
    float sv[16];
    {
      const float* sp = &Sf[wave][l15 * SF_STRIDE + hi * 16];
#pragma unroll
      for (int e = 0; e < 4; ++e) {
        v4f vq = *(const v4f*)(sp + e * 4);
#pragma unroll
        for (int c = 0; c < 4; ++c) sv[e * 4 + c] = vq[c];
      }
    }
    float mx = sv[0];
#pragma unroll
    for (int e = 1; e < 16; ++e) mx = fmaxf(mx, sv[e]);
    mx = fmaxf(mx, __shfl_xor(mx, 16, 32));       // merge the two halves
    float mnew = fmaxf(m_own, mx);
    float rescale = __expf(m_own - mnew);
    m_own = mnew;
    float psum = 0.0f;
    v8h ph[2];
#pragma unroll
    for (int e = 0; e < 16; ++e) {
      float p = __expf(sv[e] - mnew);
      psum += p;
      ph[e >> 3][e & 7] = (_Float16)p;
    }
    psum += __shfl_xor(psum, 16, 32);
    l_own = l_own * rescale + psum;
    // write P (f16) for this row half
    {
      _Float16* pw = &Pt[wave][l15 * PT_STRIDE + hi * 16];
      *(v8h*)(pw)     = ph[0];
      *(v8h*)(pw + 8) = ph[1];
    }

    // ---- rescale accumulators: broadcast per-row factor into C layout
#pragma unroll
    for (int r = 0; r < 8; ++r) {
      float s_r = __shfl(rescale, r + hi * 8, 32);  // row owner lane
#pragma unroll
      for (int t = 0; t < 4; ++t) acc[t][r] *= s_r;
    }

    asm volatile("s_wait_dscnt 0x0" ::: "memory");

    // ---- reload P in A-fragment layout (16 rows x 32 keys)
    union { v16h v; v8h h2[2]; } pf;
    const _Float16* pp = &Pt[wave][l15 * PT_STRIDE + hi * 16];
    pf.h2[0] = *(const v8h*)(pp);
    pf.h2[1] = *(const v8h*)(pp + 8);

    // ---- acc += P @ V   (4 dk N-tiles, contraction over 32 keys)
#pragma unroll
    for (int t = 0; t < 4; ++t) {
      v16h vf = *(const v16h*)(&Vt[ib][(t * 16 + l15) * VT_STRIDE + hi * 16]);
      acc[t] = wmma_f16(pf.v, vf, acc[t]);
    }
  }

  // ---- epilogue: normalize (per-row 1/l broadcast) and write f16 context
#pragma unroll
  for (int r = 0; r < 8; ++r) {
    float lr = __shfl(l_own, r + hi * 8, 32);
    float inv = 1.0f / lr;
    int row = qb * 64 + wave * 16 + r + hi * 8;
#pragma unroll
    for (int t = 0; t < 4; ++t)
      Ctx[(bS + row) * D_ + hoff + t * 16 + l15] = (_Float16)(acc[t][r] * inv);
  }
}

// ---------------------------------------------------------------------------
// Launch: convert -> transpose weights -> 3 projections -> attention -> O-proj
// ---------------------------------------------------------------------------
extern "C" void kernel_launch(void* const* d_in, const int* in_sizes, int n_in,
                              void* d_out, int out_size, void* d_ws, size_t ws_size,
                              hipStream_t stream) {
  const float* q  = (const float*)d_in[0];
  const float* k  = (const float*)d_in[1];
  const float* v  = (const float*)d_in[2];
  // d_in[3] = mask (causal, handled analytically)
  const float* wq = (const float*)d_in[4];
  const float* bq = (const float*)d_in[5];
  const float* wk = (const float*)d_in[6];
  const float* bk = (const float*)d_in[7];
  const float* wv = (const float*)d_in[8];
  const float* bv = (const float*)d_in[9];
  const float* wo = (const float*)d_in[10];
  const float* bo = (const float*)d_in[11];

  char* ws = (char*)d_ws;
  const size_t sz_x = (size_t)M_ * D_ * sizeof(_Float16);   // 16 MB
  const size_t sz_w = (size_t)D_ * D_ * sizeof(_Float16);   //  2 MB
  _Float16* xq  = (_Float16*)(ws);
  _Float16* xk  = (_Float16*)(ws + sz_x);
  _Float16* xv  = (_Float16*)(ws + 2 * sz_x);
  _Float16* wqT = (_Float16*)(ws + 3 * sz_x);
  _Float16* wkT = (_Float16*)(ws + 3 * sz_x + sz_w);
  _Float16* wvT = (_Float16*)(ws + 3 * sz_x + 2 * sz_w);
  _Float16* woT = (_Float16*)(ws + 3 * sz_x + 3 * sz_w);
  _Float16* Qh  = (_Float16*)(ws + 3 * sz_x + 4 * sz_w);
  _Float16* Kh  = (_Float16*)(ws + 4 * sz_x + 4 * sz_w);
  _Float16* Vh  = (_Float16*)(ws + 5 * sz_x + 4 * sz_w);
  _Float16* Ctx = (_Float16*)(ws + 6 * sz_x + 4 * sz_w);

  const int n  = M_ * D_;
  const int g4 = (n / 4 + 255) / 256;
  k_cvt_f16<<<g4, 256, 0, stream>>>(q, xq, n / 4);
  k_cvt_f16<<<g4, 256, 0, stream>>>(k, xk, n / 4);
  k_cvt_f16<<<g4, 256, 0, stream>>>(v, xv, n / 4);

  dim3 tg(D_ / 32, D_ / 32), tb(32, 8);
  k_wtrans<<<tg, tb, 0, stream>>>(wq, wqT);
  k_wtrans<<<tg, tb, 0, stream>>>(wk, wkT);
  k_wtrans<<<tg, tb, 0, stream>>>(wv, wvT);
  k_wtrans<<<tg, tb, 0, stream>>>(wo, woT);

  dim3 gg(D_ / 64, M_ / 128);   // (16, 64)
  k_gemm<true><<<gg, 256, 0, stream>>>(xq, wqT, bq, Qh);
  k_gemm<true><<<gg, 256, 0, stream>>>(xk, wkT, bk, Kh);
  k_gemm<true><<<gg, 256, 0, stream>>>(xv, wvT, bv, Vh);

  k_attn<<<dim3(B_ * H_, S_ / 64), 128, 0, stream>>>(Qh, Kh, Vh, Ctx);

  k_gemm<false><<<gg, 256, 0, stream>>>(Ctx, woT, bo, d_out);
}
